// GlobalLocalAttention_65781719105805
// MI455X (gfx1250) — matrix operand order repaired and checked by reading, not compiled, not run on hardware
//
#include <hip/hip_runtime.h>
#include <hip/hip_bf16.h>
#include <stdint.h>

// ---------------------------------------------------------------------------
// MI455X / gfx1250 GlobalLocalAttention forward.
// - All GEMM work on v_wmma_f32_16x16x32_bf16 (wave32)
// - LDS staging via TDM (tensor_load_to_lds + s_wait_tensorcnt) where tiles
//   are contiguous bf16 (enabled by a one-time NCHW f32 -> NHWC bf16 pass)
// ---------------------------------------------------------------------------

typedef __attribute__((ext_vector_type(16))) __bf16 v16bf;
typedef __attribute__((ext_vector_type(8)))  float  v8f;
typedef unsigned int v4u __attribute__((ext_vector_type(4)));
typedef int          v8i __attribute__((ext_vector_type(8)));
typedef int          v4i __attribute__((ext_vector_type(4)));

union BF16x16 {
  v16bf  v;
  __bf16 e[16];
  uint4  q[2];
};

static constexpr int   Bn    = 2;
static constexpr int   Cd    = 256;
static constexpr int   Hd    = 256;
static constexpr int   Wd    = 256;
static constexpr int   HWp   = Hd * Wd;  // 65536
static constexpr float BNEPS = 1e-5f;

// ---- TDM availability (device pass only; host pass lacks the builtin) ----
#if defined(__has_builtin)
#  if __has_builtin(__builtin_amdgcn_tensor_load_to_lds)
#    define HAVE_TDM 1
#  endif
#endif
#ifndef HAVE_TDM
#  define HAVE_TDM 0
#endif

#if HAVE_TDM
// 2D tensor DMA: copy tile_d1 rows of tile_d0 bf16 elements (row stride
// `stride` elements) from global to LDS byte offset lds_off. D# bitfields
// per CDNA5 ISA ch.8 (group0: count/lds/global/type, group1: dims/strides).
__device__ inline void tdm_load_bf16_2d(unsigned lds_off, const void* gptr,
                                        unsigned tile_d0, unsigned tile_d1,
                                        unsigned stride_elems) {
  const unsigned long long ga = (unsigned long long)gptr;
  v4u g0;
  g0[0] = 1u;                                            // count=1 (valid), user mode
  g0[1] = lds_off;                                       // lds_addr (bytes)
  g0[2] = (unsigned)(ga & 0xffffffffull);                // global_addr[31:0]
  g0[3] = (unsigned)((ga >> 32) & 0x1ffffffull) | (2u << 30);  // addr[56:32] | type=2
  v8i g1;
  g1[0] = (int)(1u << 16);                               // data_size=1 -> 2 bytes
  g1[1] = (int)((tile_d0 & 0xffffu) << 16);              // tensor_dim0[15:0]
  g1[2] = (int)(((tile_d0 >> 16) & 0xffffu) | ((tile_d1 & 0xffffu) << 16));
  g1[3] = (int)(((tile_d1 >> 16) & 0xffffu) | ((tile_d0 & 0xffffu) << 16)); // tile_dim0
  g1[4] = (int)(tile_d1 & 0xffffu);                      // tile_dim1 (tile_dim2=0)
  g1[5] = (int)stride_elems;                             // tensor_dim0_stride[31:0]
  g1[6] = 0;
  g1[7] = 0;
  v4i gz = {};
#if defined(__clang_major__) && (__clang_major__ >= 23)
  v8i gz8 = {};
  __builtin_amdgcn_tensor_load_to_lds(g0, g1, gz, gz, gz8, 0);
#else
  __builtin_amdgcn_tensor_load_to_lds(g0, g1, gz, gz, 0);
#endif
}
#endif  // HAVE_TDM

// ===========================================================================
// Kernel 0a: x (B,C,H,W) f32  ->  xT (B,H,W,C) bf16
// block: 256 threads = 64 pixels x 4 channel-quarters
// ===========================================================================
__global__ __launch_bounds__(256)
void k_cvt_x(const float* __restrict__ x, __bf16* __restrict__ xT) {
  const int tid = threadIdx.x;
  const int pbase = blockIdx.x * 64;
  const int b = blockIdx.y;
  const int n = tid >> 2, k0 = (tid & 3) * 64;
  const size_t xb = (size_t)b * Cd * HWp + pbase + n;
  __bf16 tmp[64];
  for (int kk = 0; kk < 64; ++kk)
    tmp[kk] = (__bf16)x[xb + (size_t)(k0 + kk) * HWp];
  __bf16* dst = xT + ((size_t)b * HWp + pbase + n) * 256 + k0;
  for (int kk = 0; kk < 64; kk += 8)
    *(uint4*)(dst + kk) = *(const uint4*)(tmp + kk);
}

// Kernel 0b: generic f32 -> bf16 (weights)
__global__ __launch_bounds__(256)
void k_cvt_w(const float* __restrict__ src, __bf16* __restrict__ dst, int n) {
  const int i = blockIdx.x * 256 + threadIdx.x;
  if (i < n) dst[i] = (__bf16)src[i];
}

// ===========================================================================
// Kernel 1: qkv = conv1x1(x, w_qkv)   (M=768, K=256, N=pixels), bf16 output
// grid: (HW/64, 768/128, B)  block: 256 (8 waves, wave = 16M x 64N)
// LDS: Wt[128][256] bf16 @0 (64KB) + Xt[64][256] bf16 @65536 (32KB)
// Both tiles contiguous bf16 in global -> staged with TDM.
// ===========================================================================
__global__ __launch_bounds__(256)
void k_qkv_gemm(const __bf16* __restrict__ xT, const __bf16* __restrict__ wqb,
                __bf16* __restrict__ qkv) {
  extern __shared__ char smem[];
  __bf16* Wt = (__bf16*)smem;            // [128][256]
  __bf16* Xt = (__bf16*)(smem + 65536);  // [64][256]  (XT[n][k])
  const int tid  = threadIdx.x;
  const int lane = tid & 31;
  const int wave = tid >> 5;
  const int hi   = lane >> 4;
  const int l15  = lane & 15;
  const int pbase = blockIdx.x * 64;
  const int ocb   = blockIdx.y * 128;
  const int b     = blockIdx.z;

  const __bf16* wsrc = wqb + (size_t)ocb * 256;
  const __bf16* xsrc = xT + ((size_t)b * HWp + pbase) * 256;
#if HAVE_TDM
  if (wave == 0) {
    tdm_load_bf16_2d(0u,     wsrc, 32768u, 1u, 32768u);
    tdm_load_bf16_2d(65536u, xsrc, 16384u, 1u, 16384u);
    __builtin_amdgcn_s_wait_tensorcnt(0);
  }
#else
  for (int i = tid; i < 32768 / 8; i += 256)
    *(uint4*)(Wt + i * 8) = *(const uint4*)(wsrc + i * 8);
  for (int i = tid; i < 16384 / 8; i += 256)
    *(uint4*)(Xt + i * 8) = *(const uint4*)(xsrc + i * 8);
#endif
  __syncthreads();

  v8f acc[4];
  for (int t = 0; t < 4; ++t) acc[t] = {};

  for (int kb = 0; kb < 8; ++kb) {
    BF16x16 a;
    const __bf16* ap = Wt + (((wave * 16 + l15) << 8) + kb * 32 + hi * 8);
    a.q[0] = *(const uint4*)(ap);
    a.q[1] = *(const uint4*)(ap + 16);
    for (int nt = 0; nt < 4; ++nt) {
      BF16x16 bb;
      const __bf16* bp = Xt + (((nt * 16 + l15) << 8) + kb * 32 + hi * 8);
      bb.q[0] = *(const uint4*)(bp);
      bb.q[1] = *(const uint4*)(bp + 16);
      acc[nt] = __builtin_amdgcn_wmma_f32_16x16x32_bf16(
          false, a.v, false, bb.v, (short)0, acc[nt], false, false);
    }
  }

  const size_t outb = (size_t)b * 768 * HWp;
  for (int nt = 0; nt < 4; ++nt) {
    const int p = pbase + nt * 16 + l15;
    for (int e = 0; e < 8; ++e) {
      const int oc = ocb + wave * 16 + hi * 8 + e;
      qkv[outb + (size_t)oc * HWp + p] = (__bf16)acc[nt][e];
    }
  }
}

// ===========================================================================
// Kernel 2: local = bn1(conv3x3(x,w1)) + bn2(conv1x1(x,w2))
// Implicit GEMM: 9 shifted taps + 1x1 tap, K blocked by 64 channels.
// grid: (HW/64, C/64, B)  block: 256 (8 waves as 4M x 2N; wave = 16oc x 32px)
// x halo read from NHWC bf16 (wide vector copies); BN scale folded into W.
// ===========================================================================
__global__ __launch_bounds__(256)
void k_local(const __bf16* __restrict__ xT,
             const float* __restrict__ w1, const float* __restrict__ g1,
             const float* __restrict__ b1,
             const float* __restrict__ w2, const float* __restrict__ g2,
             const float* __restrict__ b2,
             float* __restrict__ out) {
  extern __shared__ char smem[];
  __bf16* W1t = (__bf16*)smem;            // [(m*9+tap)*64 + k]  72KB
  __bf16* W2t = (__bf16*)(smem + 73728);  // [m*64 + k]          8KB
  __bf16* Xh  = (__bf16*)(smem + 81920);  // [(row*66+col)*64+k] ~25KB
  const int tid  = threadIdx.x;
  const int lane = tid & 31;
  const int wave = tid >> 5;
  const int hi   = lane >> 4;
  const int l15  = lane & 15;
  const int mi   = wave >> 1;
  const int ni   = wave & 1;
  const int h   = blockIdx.x >> 2;
  const int w0  = (blockIdx.x & 3) << 6;
  const int ocb = blockIdx.y << 6;
  const int b   = blockIdx.z;
  const float invs = rsqrtf(1.f + BNEPS);

  v8f acc[2];
  acc[0] = {}; acc[1] = {};

  for (int kb = 0; kb < 4; ++kb) {
    if (kb) __syncthreads();
    for (int idx = tid; idx < 64 * 9 * 64; idx += 256) {
      const int m = idx / 576, r = idx % 576, tap = r >> 6, k = r & 63;
      const float s = g1[ocb + m] * invs;
      W1t[idx] = (__bf16)(s * w1[((size_t)(ocb + m) * 256 + (kb * 64 + k)) * 9 + tap]);
    }
    for (int idx = tid; idx < 64 * 64; idx += 256) {
      const int m = idx >> 6, k = idx & 63;
      const float s = g2[ocb + m] * invs;
      W2t[idx] = (__bf16)(s * w2[(size_t)(ocb + m) * 256 + kb * 64 + k]);
    }
    // halo tile from NHWC bf16: 3 rows x 66 cols x 64 ch, uint4 granularity
    for (int idx = tid; idx < 3 * 66 * 8; idx += 256) {
      const int row = idx / (66 * 8), r2 = idx % (66 * 8);
      const int col = r2 >> 3, q4 = r2 & 7;
      const int hh = h + row - 1, ww = w0 + col - 1;
      uint4 val = make_uint4(0u, 0u, 0u, 0u);
      if (hh >= 0 && hh < Hd && ww >= 0 && ww < Wd)
        val = *(const uint4*)(xT + ((size_t)b * HWp + hh * Wd + ww) * 256 +
                              kb * 64 + q4 * 8);
      *(uint4*)(Xh + (row * 66 + col) * 64 + q4 * 8) = val;
    }
    __syncthreads();

    for (int tap = 0; tap < 10; ++tap) {
      int dy = 0, dx = 0;
      if (tap < 9) { dy = tap / 3 - 1; dx = tap % 3 - 1; }
      for (int ks = 0; ks < 2; ++ks) {
        BF16x16 a;
        const __bf16* ap =
            (tap < 9) ? (W1t + ((mi * 16 + l15) * 9 + tap) * 64 + ks * 32 + hi * 8)
                      : (W2t + (mi * 16 + l15) * 64 + ks * 32 + hi * 8);
        a.q[0] = *(const uint4*)(ap);
        a.q[1] = *(const uint4*)(ap + 16);
        for (int ns = 0; ns < 2; ++ns) {
          const int col = ni * 32 + ns * 16 + l15 + 1 + dx;
          const __bf16* bp = Xh + ((dy + 1) * 66 + col) * 64 + ks * 32 + hi * 8;
          BF16x16 bb;
          bb.q[0] = *(const uint4*)(bp);
          bb.q[1] = *(const uint4*)(bp + 16);
          acc[ns] = __builtin_amdgcn_wmma_f32_16x16x32_bf16(
              false, a.v, false, bb.v, (short)0, acc[ns], false, false);
        }
      }
    }
  }

  for (int ns = 0; ns < 2; ++ns) {
    const int px = w0 + ni * 32 + ns * 16 + l15;
    for (int e = 0; e < 8; ++e) {
      const int oc = ocb + mi * 16 + hi * 8 + e;
      out[((size_t)b * Cd + oc) * HWp + h * Wd + px] = acc[ns][e] + b1[oc] + b2[oc];
    }
  }
}

// ===========================================================================
// Kernel 3: windowed attention (bug-faithful flat reinterpretation).
// One window per WG; 128 threads = 4 waves, 4 heads per wave.
// Window qkv tile = 8 rows x 12288B (stride 384KB): one 2D TDM descriptor.
// q@k^T : K=16 padded to 32 with zeros.  attn@v : K=64 -> 2 wmma steps.
// ===========================================================================
__global__ __launch_bounds__(128)
void k_attn(const __bf16* __restrict__ qkv, const float* __restrict__ rpb,
            float* __restrict__ attn) {
  extern __shared__ char smem[];
  __bf16* qs = (__bf16*)smem;  // [64][768]  96KB
  const int tid  = threadIdx.x;
  const int lane = tid & 31;
  const int wave = tid >> 5;
  const int hi   = lane >> 4;
  const int l15  = lane & 15;
  float*  dots  = (float*)(smem + 98304 + wave * 16384);   // [64][64] f32
  __bf16* probs = (__bf16*)(smem + 163840 + wave * 8192);  // [64][64] bf16

  const int n = blockIdx.x;
  const int b = n >> 10, rem = n & 1023, i = rem >> 5, j = rem & 31;
  // flat-reinterpreted window base: element 768*(b*HW + (8i+r)*W + 8j+s)
  const size_t base3 = (size_t)768 * ((size_t)b * HWp + (i * 8) * Wd + j * 8);

#if HAVE_TDM
  if (wave == 0) {
    tdm_load_bf16_2d(0u, qkv + base3, 6144u, 8u, (unsigned)(Wd * 768));
    __builtin_amdgcn_s_wait_tensorcnt(0);
  }
#else
  for (int idx = tid; idx < 6144; idx += 128) {
    const int r = idx / 768, c = idx % 768;
    const __bf16* src = qkv + base3 + (size_t)r * Wd * 768 + (size_t)c * 8;
    *(uint4*)(qs + r * 6144 + c * 8) = *(const uint4*)src;
  }
#endif
  __syncthreads();

  for (int hh = 0; hh < 4; ++hh) {
    const int head = wave * 4 + hh;
    // ---- dots = 0.25 * q@k^T + relpos bias ----
    for (int qt = 0; qt < 4; ++qt) {
      BF16x16 a;
      {
        const __bf16* ap = qs + (qt * 16 + l15) * 768 + head * 16 + hi * 8;
        a.q[0] = *(const uint4*)ap;
        for (int e = 8; e < 16; ++e) a.e[e] = (__bf16)0.f;  // K pad 16->32
      }
      for (int kt = 0; kt < 4; ++kt) {
        BF16x16 bb;
        {
          const __bf16* bp = qs + (kt * 16 + l15) * 768 + 256 + head * 16 + hi * 8;
          bb.q[0] = *(const uint4*)(bp);
          for (int e = 8; e < 16; ++e) bb.e[e] = (__bf16)0.f;  // K pad
        }
        v8f d = {};
        d = __builtin_amdgcn_wmma_f32_16x16x32_bf16(
            false, a.v, false, bb.v, (short)0, d, false, false);
        const int ktok = kt * 16 + l15;
        const int kr = ktok >> 3, kc = ktok & 7;
        for (int e = 0; e < 8; ++e) {
          const int qtok = qt * 16 + hi * 8 + e;
          const int qr = qtok >> 3, qc = qtok & 7;
          const float bias = rpb[((qr - kr + 7) * 15 + (qc - kc + 7)) * 16 + head];
          dots[qtok * 64 + ktok] = d[e] * 0.25f + bias;
        }
      }
    }
    // ---- softmax over k (wave-local; LDS ops in-order within a wave) ----
    for (int rr = lane; rr < 64; rr += 32) {
      float mx = -1e30f;
      for (int k = 0; k < 64; ++k) mx = fmaxf(mx, dots[rr * 64 + k]);
      float sum = 0.f;
      for (int k = 0; k < 64; ++k) sum += __expf(dots[rr * 64 + k] - mx);
      const float inv = 1.f / sum;
      for (int k = 0; k < 64; ++k)
        probs[rr * 64 + k] = (__bf16)(__expf(dots[rr * 64 + k] - mx) * inv);
    }
    // ---- out = probs @ v ----
    BF16x16 vfrag[2];
    for (int ks = 0; ks < 2; ++ks)
      for (int e = 0; e < 16; ++e) {
        const int ktok = ks * 32 + hi * 8 + (e & 7) + (e >> 3) * 16;
        vfrag[ks].e[e] = qs[ktok * 768 + 512 + head * 16 + l15];
      }
    for (int qt = 0; qt < 4; ++qt) {
      v8f o = {};
      for (int ks = 0; ks < 2; ++ks) {
        BF16x16 a2;
        const __bf16* pr = probs + (qt * 16 + l15) * 64 + ks * 32 + hi * 8;
        a2.q[0] = *(const uint4*)(pr);
        a2.q[1] = *(const uint4*)(pr + 16);
        o = __builtin_amdgcn_wmma_f32_16x16x32_bf16(
            false, a2.v, false, vfrag[ks].v, (short)0, o, false, false);
      }
      for (int e = 0; e < 8; ++e) {
        const int qtok = qt * 16 + hi * 8 + e;
        attn[(((size_t)n * 16 + head) * 64 + qtok) * 16 + l15] = o[e];
      }
    }
  }
}

// ===========================================================================
// Kernel 4: comb = avgpool_x(reflect) + avgpool_y(reflect) + local
// ===========================================================================
__global__ __launch_bounds__(256)
void k_combine(const float* __restrict__ attn, const float* __restrict__ local,
               float* __restrict__ comb) {
  const size_t idx = (size_t)blockIdx.x * blockDim.x + threadIdx.x;
  const int w = idx & 255;
  const int h = (idx >> 8) & 255;
  const size_t plane = idx >> 16;  // b*C + c
  const float* base = attn + (plane << 16);
  float sx = 0.f, sy = 0.f;
  for (int t = 0; t < 8; ++t) {
    const int r = h - 3 + t;
    if (r >= 0 && r <= 256) { const int rr = (r == 256) ? 254 : r; sx += base[rr * 256 + w]; }
    const int c = w - 3 + t;
    if (c >= 0 && c <= 256) { const int cc = (c == 256) ? 254 : c; sy += base[h * 256 + cc]; }
  }
  comb[idx] = (sx + sy) * 0.125f + local[idx];
}

// ===========================================================================
// Kernel 5: depthwise 8x8 (reflect (0,1),(0,1) then zero-pad 3) + BN3
// Output written as NHWC bf16 so the final GEMM can TDM-load it.
// grid: (64 tiles, C, B)  block: 256; 32x32 output tile, 39x39 halo in LDS.
// ===========================================================================
__global__ __launch_bounds__(256)
void k_dwconv(const float* __restrict__ comb, const float* __restrict__ wdw,
              const float* __restrict__ g3, const float* __restrict__ b3,
              __bf16* __restrict__ dwT) {
  extern __shared__ float sm[];
  float* tile = sm;            // [39][40]
  float* wt   = sm + 39 * 40;  // [64]
  const int tid = threadIdx.x;
  const int c = blockIdx.y, b = blockIdx.z;
  const int h0 = (blockIdx.x >> 3) * 32, w0 = (blockIdx.x & 7) * 32;
  const float* src = comb + (((size_t)b * Cd + c) << 16);
  for (int i = tid; i < 39 * 39; i += 256) {
    const int r = i / 39, cc = i % 39;
    const int hh = h0 - 3 + r, ww = w0 - 3 + cc;
    float v = 0.f;
    if (hh >= 0 && hh <= 256 && ww >= 0 && ww <= 256) {
      const int hr = (hh == 256) ? 254 : hh;
      const int wr = (ww == 256) ? 254 : ww;
      v = src[hr * 256 + wr];
    }
    tile[r * 40 + cc] = v;
  }
  if (tid < 64) wt[tid] = wdw[c * 64 + tid];
  __syncthreads();
  const float s  = g3[c] * rsqrtf(1.f + BNEPS);
  const float bb = b3[c];
  for (int q = 0; q < 4; ++q) {
    const int pix = tid + q * 256;
    const int oy = pix >> 5, ox = pix & 31;
    float acc = 0.f;
    for (int dy = 0; dy < 8; ++dy)
      for (int dx = 0; dx < 8; ++dx)
        acc += wt[dy * 8 + dx] * tile[(oy + dy) * 40 + ox + dx];
    dwT[((size_t)b * HWp + (h0 + oy) * Wd + (w0 + ox)) * 256 + c] =
        (__bf16)(acc * s + bb);
  }
}

// ===========================================================================
// Kernel 6: out = conv1x1(dw, w_pw)  (M=256, K=256), f32 -> d_out
// Both LDS tiles contiguous bf16 in global -> TDM staged.
// ===========================================================================
__global__ __launch_bounds__(256)
void k_pw(const __bf16* __restrict__ dwT, const __bf16* __restrict__ wpb,
          float* __restrict__ out) {
  extern __shared__ char smem[];
  __bf16* Wt = (__bf16*)smem;            // [128][256]
  __bf16* Xt = (__bf16*)(smem + 65536);  // [64][256]
  const int tid  = threadIdx.x;
  const int lane = tid & 31;
  const int wave = tid >> 5;
  const int hi   = lane >> 4;
  const int l15  = lane & 15;
  const int pbase = blockIdx.x * 64;
  const int ocb   = blockIdx.y * 128;
  const int b     = blockIdx.z;

  const __bf16* wsrc = wpb + (size_t)ocb * 256;
  const __bf16* xsrc = dwT + ((size_t)b * HWp + pbase) * 256;
#if HAVE_TDM
  if (wave == 0) {
    tdm_load_bf16_2d(0u,     wsrc, 32768u, 1u, 32768u);
    tdm_load_bf16_2d(65536u, xsrc, 16384u, 1u, 16384u);
    __builtin_amdgcn_s_wait_tensorcnt(0);
  }
#else
  for (int i = tid; i < 32768 / 8; i += 256)
    *(uint4*)(Wt + i * 8) = *(const uint4*)(wsrc + i * 8);
  for (int i = tid; i < 16384 / 8; i += 256)
    *(uint4*)(Xt + i * 8) = *(const uint4*)(xsrc + i * 8);
#endif
  __syncthreads();

  v8f acc[4];
  for (int t = 0; t < 4; ++t) acc[t] = {};
  for (int kb = 0; kb < 8; ++kb) {
    BF16x16 a;
    const __bf16* ap = Wt + (((wave * 16 + l15) << 8) + kb * 32 + hi * 8);
    a.q[0] = *(const uint4*)(ap);
    a.q[1] = *(const uint4*)(ap + 16);
    for (int nt = 0; nt < 4; ++nt) {
      BF16x16 bb;
      const __bf16* bp = Xt + (((nt * 16 + l15) << 8) + kb * 32 + hi * 8);
      bb.q[0] = *(const uint4*)(bp);
      bb.q[1] = *(const uint4*)(bp + 16);
      acc[nt] = __builtin_amdgcn_wmma_f32_16x16x32_bf16(
          false, a.v, false, bb.v, (short)0, acc[nt], false, false);
    }
  }
  const size_t outb = (size_t)b * Cd * HWp;
  for (int nt = 0; nt < 4; ++nt) {
    const int p = pbase + nt * 16 + l15;
    for (int e = 0; e < 8; ++e) {
      const int oc = ocb + wave * 16 + hi * 8 + e;
      out[outb + (size_t)oc * HWp + p] = acc[nt][e];
    }
  }
}

// ===========================================================================
extern "C" void kernel_launch(void* const* d_in, const int* in_sizes, int n_in,
                              void* d_out, int out_size, void* d_ws, size_t ws_size,
                              hipStream_t stream) {
  (void)in_sizes; (void)n_in; (void)out_size; (void)ws_size;
  const float* x   = (const float*)d_in[0];
  const float* wq  = (const float*)d_in[1];
  const float* w1  = (const float*)d_in[2];
  const float* g1  = (const float*)d_in[3];
  const float* b1  = (const float*)d_in[4];
  const float* w2  = (const float*)d_in[5];
  const float* g2  = (const float*)d_in[6];
  const float* b2  = (const float*)d_in[7];
  const float* wdw = (const float*)d_in[8];
  const float* g3  = (const float*)d_in[9];
  const float* b3  = (const float*)d_in[10];
  const float* wp  = (const float*)d_in[11];
  const float* rpb = (const float*)d_in[12];
  float* out = (float*)d_out;

  char* ws = (char*)d_ws;
  __bf16* qkv   = (__bf16*)ws;                     // 201,326,592 B
  float*  local = (float*)(ws + 201326592ull);     // 134,217,728 B
  float*  attn  = (float*)(ws + 335544320ull);     // 134,217,728 B
  __bf16* xT    = (__bf16*)(ws + 469762048ull);    //  67,108,864 B (NHWC bf16)
  __bf16* wqb   = (__bf16*)(ws + 536870912ull);    //     393,216 B
  __bf16* wpb   = (__bf16*)(ws + 537264128ull);    //     131,072 B
  float*  comb  = (float*)ws;                      // reuse (qkv dead)
  __bf16* dwT   = (__bf16*)(ws + 201326592ull);    // reuse (local dead), NHWC bf16

  k_cvt_x  <<<dim3(HWp / 64, Bn), 256, 0, stream>>>(x, xT);
  k_cvt_w  <<<dim3((768 * 256 + 255) / 256), 256, 0, stream>>>(wq, wqb, 768 * 256);
  k_cvt_w  <<<dim3((256 * 256 + 255) / 256), 256, 0, stream>>>(wp, wpb, 256 * 256);

  k_qkv_gemm<<<dim3(HWp / 64, 6, Bn), 256, 98304, stream>>>(xT, wqb, qkv);
  k_local  <<<dim3(HWp / 64, 4, Bn), 256, 107264, stream>>>(xT, w1, g1, b1, w2, g2, b2, local);
  k_attn   <<<dim3(2048), 128, 196608, stream>>>(qkv, rpb, attn);
  k_combine<<<dim3((Bn * Cd * HWp) / 256), 256, 0, stream>>>(attn, local, comb);
  k_dwconv <<<dim3(64, Cd, Bn), 256, (39 * 40 + 64) * 4, stream>>>(comb, wdw, g3, b3, dwT);
  k_pw     <<<dim3(HWp / 64, 2, Bn), 256, 98304, stream>>>(dwT, wpb, out);
}
